// SBF_nlfc_34754875359797
// MI455X (gfx1250) — compile-verified, matrix operand order checked by reading, not run on hardware
//
#include <hip/hip_runtime.h>
#include <hip/hip_bf16.h>

typedef __attribute__((ext_vector_type(16))) _Float16 v16h;
typedef __attribute__((ext_vector_type(8)))  _Float16 v8h;
typedef __attribute__((ext_vector_type(8)))  float    v8f;

#define BB 32
#define CC 1024
#define CI 512
#define NN 784
#define NP 800   // N padded to a multiple of 32 for WMMA K-loops / B-operand rows

// Load one 16x32 f16 WMMA operand fragment (A or B) from a row-major [rows x K]
// matrix. Per ISA layout: lane%16 = row, lane/16 selects the 8-wide K-subchunk;
// vector elements [0:8] = K kb..kb+7, [8:16] = K kb+16..kb+23, kb=(lane>>4)*8.
// Caller bakes row*ld + kb into the pointer; here we just grab the two 16B chunks.
__device__ __forceinline__ v16h ld_frag(const _Float16* p) {
  union { v16h v; v8h h[2]; } u;
  u.h[0] = *reinterpret_cast<const v8h*>(p);
  u.h[1] = *reinterpret_cast<const v8h*>(p + 16);
  return u.v;
}

#define WMMA(a, b, c) __builtin_amdgcn_wmma_f32_16x16x32_f16(false, (a), false, (b), (short)0, (c), false, false)

// ---------- data prep ----------

// xh_t[b][n][c] = f16(x[b][c][n])  (read coalesced in n, scattered f16 writes)
__global__ __launch_bounds__(256) void k_cvt_x(const float* __restrict__ x,
                                               _Float16* __restrict__ xh_t) {
  int idx = blockIdx.x * 256 + threadIdx.x;
  int n = idx % NN;
  int t = idx / NN;
  int c = t % CC;
  int b = t / CC;
  xh_t[((size_t)b * NN + n) * CC + c] = (_Float16)x[idx];
}

// xmean[b][c] = mean_n x[b][c][n]  (one wave per row)
__global__ __launch_bounds__(256) void k_xmean(const float* __restrict__ x,
                                               float* __restrict__ xmean) {
  int lane = threadIdx.x & 31;
  int row = blockIdx.x * 8 + (threadIdx.x >> 5);   // b*C + c
  const float* src = x + (size_t)row * NN;
  float s = 0.f;
  for (int n = lane; n < NN; n += 32) s += src[n];
  for (int off = 16; off; off >>= 1) s += __shfl_xor(s, off, 32);
  if (lane == 0) xmean[row] = s * (1.0f / NN);
}

// wallh = f16([Wth; Wph; Wg])  (1536x1024), wwh = f16(Ww) (1024x512)
__global__ __launch_bounds__(256) void k_cvt_w(const float* __restrict__ Wth,
                                               const float* __restrict__ Wph,
                                               const float* __restrict__ Wg,
                                               const float* __restrict__ Ww,
                                               _Float16* __restrict__ wallh,
                                               _Float16* __restrict__ wwh) {
  int idx = blockIdx.x * 256 + threadIdx.x;
  if (idx < 1536 * 1024) {
    int i = idx >> 10;
    float v = (i < 512) ? Wth[idx]
            : (i < 1024) ? Wph[idx - 512 * 1024]
                         : Wg[idx - 1024 * 1024];
    wallh[idx] = (_Float16)v;
  } else {
    int j = idx - 1536 * 1024;
    wwh[j] = (_Float16)Ww[j];
  }
}

// ---------- WMMA GEMMs: one 16(M)x32(N) D tile per wave (1x2 register
// blocking: one A frag feeds two B frags -> 3 b128 loads per WMMA) ----------

// D[n, j] = sum_c xh_t[b][n][c] * wall[j][c];  j<512 -> theta_t[b][n][j]+bth,
// 512..1023 -> phi_t[b][n][j-512]+bph (800-row pad), 1024.. -> g[b][j-1024][n]+bg
__global__ __launch_bounds__(256) void k_proj(const _Float16* __restrict__ xh_t,
                                              const _Float16* __restrict__ wallh,
                                              const float* __restrict__ bth,
                                              const float* __restrict__ bph,
                                              const float* __restrict__ bg,
                                              _Float16* __restrict__ theta_t,
                                              _Float16* __restrict__ phi_t,
                                              _Float16* __restrict__ g) {
  int lane = threadIdx.x & 31;
  int wid = blockIdx.x * 8 + (threadIdx.x >> 5);
  int b  = wid / (49 * 48);
  int r  = wid % (49 * 48);
  int mt = r / 48, nt = r % 48;                  // nt = 32-wide column pair
  const _Float16* pa  = xh_t + (size_t)b * NN * CC
                      + (size_t)(mt * 16 + (lane & 15)) * CC + ((lane >> 4) << 3);
  const _Float16* pb0 = wallh
                      + (size_t)(nt * 32 + (lane & 15)) * CC + ((lane >> 4) << 3);
  const _Float16* pb1 = pb0 + (size_t)16 * CC;
  v8f acc0 = {}, acc1 = {};
  for (int kk = 0; kk < CC; kk += 32) {
    __builtin_prefetch(pa  + kk + 128, 0, 3);
    __builtin_prefetch(pb0 + kk + 128, 0, 3);
    __builtin_prefetch(pb1 + kk + 128, 0, 3);
    v16h a  = ld_frag(pa  + kk);
    v16h b0 = ld_frag(pb0 + kk);
    v16h b1 = ld_frag(pb1 + kk);
    acc0 = WMMA(a, b0, acc0);
    acc1 = WMMA(a, b1, acc1);
  }
  int j0 = nt * 32 + (lane & 15);
  int j1 = j0 + 16;                              // same 512-segment as j0
  int rowbase = mt * 16 + ((lane >> 4) << 3);    // n index base
  if (j0 < 512) {
    float bi0 = bth[j0], bi1 = bth[j1];
    for (int rr = 0; rr < 8; ++rr) {
      size_t ro = ((size_t)b * NN + rowbase + rr) * CI;
      theta_t[ro + j0] = (_Float16)(acc0[rr] + bi0);
      theta_t[ro + j1] = (_Float16)(acc1[rr] + bi1);
    }
  } else if (j0 < 1024) {
    int jj0 = j0 - 512, jj1 = j1 - 512;
    float bi0 = bph[jj0], bi1 = bph[jj1];
    for (int rr = 0; rr < 8; ++rr) {
      size_t ro = ((size_t)b * NP + rowbase + rr) * CI;   // phi padded to 800 rows
      phi_t[ro + jj0] = (_Float16)(acc0[rr] + bi0);
      phi_t[ro + jj1] = (_Float16)(acc1[rr] + bi1);
    }
  } else {
    int jj0 = j0 - 1024, jj1 = j1 - 1024;
    float bi0 = bg[jj0], bi1 = bg[jj1];
    for (int rr = 0; rr < 8; ++rr) {
      g[((size_t)b * CI + jj0) * NP + rowbase + rr] = (_Float16)(acc0[rr] + bi0);
      g[((size_t)b * CI + jj1) * NP + rowbase + rr] = (_Float16)(acc1[rr] + bi1);
    }
  }
}

// scores[b][n][m] = sum_i theta_t[b][n][i] * phi_t[b][m][i]  (scores stride NP;
// pad columns m>=784 are written but never read by softmax)
__global__ __launch_bounds__(256) void k_scores(const _Float16* __restrict__ theta_t,
                                                const _Float16* __restrict__ phi_t,
                                                float* __restrict__ scores) {
  int lane = threadIdx.x & 31;
  int wid = blockIdx.x * 8 + (threadIdx.x >> 5);
  int b  = wid / (49 * 25);
  int r  = wid % (49 * 25);
  int mt = r / 25, nt = r % 25;
  const _Float16* pa  = theta_t + (size_t)b * NN * CI
                      + (size_t)(mt * 16 + (lane & 15)) * CI + ((lane >> 4) << 3);
  const _Float16* pb0 = phi_t + (size_t)b * NP * CI
                      + (size_t)(nt * 32 + (lane & 15)) * CI + ((lane >> 4) << 3);
  const _Float16* pb1 = pb0 + (size_t)16 * CI;
  v8f acc0 = {}, acc1 = {};
  for (int kk = 0; kk < CI; kk += 32) {
    __builtin_prefetch(pa  + kk + 128, 0, 3);
    __builtin_prefetch(pb0 + kk + 128, 0, 3);
    __builtin_prefetch(pb1 + kk + 128, 0, 3);
    v16h a  = ld_frag(pa  + kk);
    v16h b0 = ld_frag(pb0 + kk);
    v16h b1 = ld_frag(pb1 + kk);
    acc0 = WMMA(a, b0, acc0);
    acc1 = WMMA(a, b1, acc1);
  }
  float* out = scores + (size_t)b * NN * NP;
  int m0 = nt * 32 + (lane & 15);
  int rowbase = mt * 16 + ((lane >> 4) << 3);
  for (int rr = 0; rr < 8; ++rr) {
    size_t ro = (size_t)(rowbase + rr) * NP;
    out[ro + m0]      = acc0[rr];
    out[ro + m0 + 16] = acc1[rr];
  }
}

// row softmax over m=784; writes f16 attn with K padded to 800 (pad zeroed)
__global__ __launch_bounds__(256) void k_softmax(const float* __restrict__ scores,
                                                 _Float16* __restrict__ attn) {
  __shared__ float sred[8];
  __shared__ float sred2[8];
  int row = blockIdx.x;                      // b*784 + n
  const float* src = scores + (size_t)row * NP;
  _Float16* dst = attn + (size_t)row * NP;
  int tid = threadIdx.x, lane = tid & 31, w = tid >> 5;
  float m = -3.0e38f;
  for (int i = tid; i < NN; i += 256) m = fmaxf(m, src[i]);
  for (int off = 16; off; off >>= 1) m = fmaxf(m, __shfl_xor(m, off, 32));
  if (lane == 0) sred[w] = m;
  __syncthreads();
  float rmax = fmaxf(fmaxf(fmaxf(sred[0], sred[1]), fmaxf(sred[2], sred[3])),
                     fmaxf(fmaxf(sred[4], sred[5]), fmaxf(sred[6], sred[7])));
  float s = 0.f;
  for (int i = tid; i < NN; i += 256) s += __expf(src[i] - rmax);
  for (int off = 16; off; off >>= 1) s += __shfl_xor(s, off, 32);
  if (lane == 0) sred2[w] = s;
  __syncthreads();
  float rsum = sred2[0] + sred2[1] + sred2[2] + sred2[3]
             + sred2[4] + sred2[5] + sred2[6] + sred2[7];
  float inv = 1.0f / rsum;
  for (int i = tid; i < NN; i += 256) dst[i] = (_Float16)(__expf(src[i] - rmax) * inv);
  if (tid < NP - NN) dst[NN + tid] = (_Float16)0.0f;
}

// y_t[b][n][i] = sum_m attn[b][n][m] * g[b][i][m]   (K = NP, zero padded)
__global__ __launch_bounds__(256) void k_y(const _Float16* __restrict__ attn,
                                           const _Float16* __restrict__ g,
                                           _Float16* __restrict__ y_t) {
  int lane = threadIdx.x & 31;
  int wid = blockIdx.x * 8 + (threadIdx.x >> 5);
  int b  = wid / (49 * 16);
  int r  = wid % (49 * 16);
  int mt = r / 16, nt = r % 16;
  const _Float16* pa  = attn + (size_t)b * NN * NP
                      + (size_t)(mt * 16 + (lane & 15)) * NP + ((lane >> 4) << 3);
  const _Float16* pb0 = g + (size_t)b * CI * NP
                      + (size_t)(nt * 32 + (lane & 15)) * NP + ((lane >> 4) << 3);
  const _Float16* pb1 = pb0 + (size_t)16 * NP;
  v8f acc0 = {}, acc1 = {};
  for (int kk = 0; kk < NP; kk += 32) {
    __builtin_prefetch(pa  + kk + 128, 0, 3);
    __builtin_prefetch(pb0 + kk + 128, 0, 3);
    __builtin_prefetch(pb1 + kk + 128, 0, 3);
    v16h a  = ld_frag(pa  + kk);
    v16h b0 = ld_frag(pb0 + kk);
    v16h b1 = ld_frag(pb1 + kk);
    acc0 = WMMA(a, b0, acc0);
    acc1 = WMMA(a, b1, acc1);
  }
  int i0 = nt * 32 + (lane & 15);
  int rowbase = mt * 16 + ((lane >> 4) << 3);
  for (int rr = 0; rr < 8; ++rr) {
    size_t ro = ((size_t)b * NN + rowbase + rr) * CI;
    y_t[ro + i0]      = (_Float16)acc0[rr];
    y_t[ro + i0 + 16] = (_Float16)acc1[rr];
  }
}

// D[n][c] = sum_i y_t[b][n][i]*Ww[c][i]; accumulate sum_n D[n][c] into pacc[b][c]
__global__ __launch_bounds__(256) void k_z(const _Float16* __restrict__ y_t,
                                           const _Float16* __restrict__ wwh,
                                           float* __restrict__ pacc) {
  int lane = threadIdx.x & 31;
  int wid = blockIdx.x * 8 + (threadIdx.x >> 5);
  int b  = wid / (49 * 32);
  int r  = wid % (49 * 32);
  int mt = r / 32, nt = r % 32;
  const _Float16* pa  = y_t + (size_t)b * NN * CI
                      + (size_t)(mt * 16 + (lane & 15)) * CI + ((lane >> 4) << 3);
  const _Float16* pb0 = wwh
                      + (size_t)(nt * 32 + (lane & 15)) * CI + ((lane >> 4) << 3);
  const _Float16* pb1 = pb0 + (size_t)16 * CI;
  v8f acc0 = {}, acc1 = {};
  for (int kk = 0; kk < CI; kk += 32) {
    __builtin_prefetch(pa + kk + 128, 0, 3);
    v16h a  = ld_frag(pa  + kk);
    v16h b0 = ld_frag(pb0 + kk);
    v16h b1 = ld_frag(pb1 + kk);
    acc0 = WMMA(a, b0, acc0);
    acc1 = WMMA(a, b1, acc1);
  }
  float s0 = acc0[0] + acc0[1] + acc0[2] + acc0[3] + acc0[4] + acc0[5] + acc0[6] + acc0[7];
  float s1 = acc1[0] + acc1[1] + acc1[2] + acc1[3] + acc1[4] + acc1[5] + acc1[6] + acc1[7];
  s0 += __shfl_xor(s0, 16, 32);               // lanes l and l+16 share column c
  s1 += __shfl_xor(s1, 16, 32);
  if (lane < 16) {
    int c = nt * 32 + lane;
    atomicAdd(&pacc[b * CC + c], s0);
    atomicAdd(&pacc[b * CC + c + 16], s1);
  }
}

// pooled[b][c] = pacc/784*inv + bw*inv + beta - mu*inv + xmean  (BN folded)
__global__ __launch_bounds__(256) void k_final(const float* __restrict__ pacc,
                                               const float* __restrict__ xmean,
                                               const float* __restrict__ bw,
                                               const float* __restrict__ gam,
                                               const float* __restrict__ bet,
                                               const float* __restrict__ mu,
                                               const float* __restrict__ var,
                                               float* __restrict__ pfinal) {
  int idx = blockIdx.x * 256 + threadIdx.x;   // b*C + c
  int c = idx & (CC - 1);
  float inv = gam[c] * rsqrtf(var[c] + 1e-5f);
  pfinal[idx] = pacc[idx] * (1.0f / NN) * inv + bw[c] * inv + bet[c] - mu[c] * inv
              + xmean[idx];
}

// h[b][j] = b1[j] + pooled[b] . W1[j]
__global__ __launch_bounds__(256) void k_fc1(const float* __restrict__ pfinal,
                                             const float* __restrict__ W1,
                                             const float* __restrict__ b1,
                                             float* __restrict__ h) {
  int idx = blockIdx.x * 256 + threadIdx.x;   // b*512 + j
  int j = idx & 511, b = idx >> 9;
  const float* p = pfinal + b * CC;
  const float* w = W1 + (size_t)j * CC;
  float s = b1[j];
  for (int c = 0; c < CC; ++c) s += p[c] * w[c];
  h[idx] = s;
}

// out[b][o] = b2[o] + [h | x_face] . W2[o]
__global__ void k_fc2(const float* __restrict__ h, const float* __restrict__ xface,
                      const float* __restrict__ W2, const float* __restrict__ b2,
                      float* __restrict__ out) {
  int idx = threadIdx.x;                      // 0..63
  int b = idx >> 1, o = idx & 1;
  const float* w  = W2 + o * 1024;
  const float* hb = h + b * 512;
  const float* fb = xface + b * 512;
  float s = b2[o];
  for (int j = 0; j < 512; ++j) s += hb[j] * w[j];
  for (int j = 0; j < 512; ++j) s += fb[j] * w[512 + j];
  out[b * 2 + o] = s;
}

extern "C" void kernel_launch(void* const* d_in, const int* in_sizes, int n_in,
                              void* d_out, int out_size, void* d_ws, size_t ws_size,
                              hipStream_t stream) {
  const float* x_body = (const float*)d_in[0];
  const float* x_face = (const float*)d_in[1];
  const float* Wg  = (const float*)d_in[2];
  const float* bg  = (const float*)d_in[3];
  const float* Wth = (const float*)d_in[4];
  const float* bth = (const float*)d_in[5];
  const float* Wph = (const float*)d_in[6];
  const float* bph = (const float*)d_in[7];
  const float* Ww  = (const float*)d_in[8];
  const float* bw  = (const float*)d_in[9];
  const float* gam = (const float*)d_in[10];
  const float* bet = (const float*)d_in[11];
  const float* mu  = (const float*)d_in[12];
  const float* var = (const float*)d_in[13];
  const float* W1  = (const float*)d_in[14];
  const float* b1  = (const float*)d_in[15];
  const float* W2  = (const float*)d_in[16];
  const float* b2  = (const float*)d_in[17];
  float* out = (float*)d_out;

  // workspace carve-up (all region sizes are multiples of 256 bytes)
  char* ws = (char*)d_ws;
  size_t off = 0;
  _Float16* xh_t   = (_Float16*)(ws + off); off += (size_t)BB * NN * CC * 2;  // 51.4 MB
  _Float16* wallh  = (_Float16*)(ws + off); off += (size_t)1536 * CC * 2;     //  3.1 MB
  _Float16* wwh    = (_Float16*)(ws + off); off += (size_t)CC * CI * 2;       //  1.0 MB
  _Float16* thetaT = (_Float16*)(ws + off); off += (size_t)BB * NN * CI * 2;  // 25.7 MB
  _Float16* phiT   = (_Float16*)(ws + off); off += (size_t)BB * NP * CI * 2;  // 26.2 MB (800 rows)
  _Float16* gbuf   = (_Float16*)(ws + off); size_t gBytes = (size_t)BB * CI * NP * 2;
  off += gBytes;                                                              // 26.2 MB
  float*    scores = (float*)(ws + off);    off += (size_t)BB * NN * NP * 4;  // 80.3 MB (stride 800)
  _Float16* attn   = (_Float16*)(ws + off); off += (size_t)BB * NN * NP * 2;  // 40.1 MB
  _Float16* yT     = (_Float16*)(ws + off); off += (size_t)BB * NN * CI * 2;  // 25.7 MB
  float*    xmean  = (float*)(ws + off);    off += (size_t)BB * CC * 4;
  float*    pacc   = (float*)(ws + off);    off += (size_t)BB * CC * 4;
  float*    pfinal = (float*)(ws + off);    off += (size_t)BB * CC * 4;
  float*    hbuf   = (float*)(ws + off);    off += (size_t)BB * 512 * 4;
  (void)off; (void)ws_size; (void)in_sizes; (void)n_in; (void)out_size;

  // zero the g buffer (so its K padding contributes 0) and the pooled accumulator
  hipMemsetAsync(gbuf, 0, gBytes, stream);
  hipMemsetAsync(pacc, 0, (size_t)BB * CC * 4, stream);

  k_cvt_x  <<<100352, 256, 0, stream>>>(x_body, xh_t);
  k_xmean  <<<4096,   256, 0, stream>>>(x_body, xmean);
  k_cvt_w  <<<8192,   256, 0, stream>>>(Wth, Wph, Wg, Ww, wallh, wwh);
  k_proj   <<<9408,   256, 0, stream>>>(xh_t, wallh, bth, bph, bg, thetaT, phiT, gbuf);
  k_scores <<<4900,   256, 0, stream>>>(thetaT, phiT, scores);
  k_softmax<<<BB * NN,256, 0, stream>>>(scores, attn);
  k_y      <<<3136,   256, 0, stream>>>(attn, gbuf, yT);
  k_z      <<<6272,   256, 0, stream>>>(yT, wwh, pacc);
  k_final  <<<128,    256, 0, stream>>>(pacc, xmean, bw, gam, bet, mu, var, pfinal);
  k_fc1    <<<64,     256, 0, stream>>>(pfinal, W1, b1, hbuf);
  k_fc2    <<<1,      64,  0, stream>>>(hbuf, x_face, W2, b2, out);
}
